// SelfAttention_8675833938254
// MI455X (gfx1250) — matrix operand order repaired and checked by reading, not compiled, not run on hardware
//
#include <hip/hip_runtime.h>

typedef __attribute__((ext_vector_type(16))) __bf16 v16bf;
typedef __attribute__((ext_vector_type(8)))  __bf16 v8bf;
typedef __attribute__((ext_vector_type(8)))  float  v8f;
typedef unsigned int v4u __attribute__((ext_vector_type(4)));
typedef int v8i __attribute__((ext_vector_type(8)));
typedef int v4i __attribute__((ext_vector_type(4)));

#define BT    2048
#define CDIM  1024
#define NH    16
#define HD    64
#define BATCH 4
#define MTOT  (BATCH * BT)   // 8192

__device__ __forceinline__ unsigned short f2bf(float f) {
  union { float f; unsigned u; } v; v.f = f;
  unsigned u = v.u;
  u += 0x7FFFu + ((u >> 16) & 1u);
  return (unsigned short)(u >> 16);
}

// packed f32x2 -> bf16x2 (one v_cvt_pk_bf16_f32 when available)
__device__ __forceinline__ unsigned pkbf(float a, float b) {
#if __has_builtin(__builtin_amdgcn_cvt_pk_bf16_f32)
  typedef __attribute__((ext_vector_type(2))) __bf16 v2bf;
  v2bf r = __builtin_amdgcn_cvt_pk_bf16_f32(a, b);
  return __builtin_bit_cast(unsigned, r);
#else
  return (unsigned)f2bf(a) | ((unsigned)f2bf(b) << 16);
#endif
}

// ---------------------------------------------------------------------------
// TDM: 2D bf16 tile load, global -> LDS, with LDS row padding of 16B per
// 128B row (rows of 64 bf16 land in [..][72]-short padded LDS rows).
// tile_d0 must be 64 (one 128B row = 32 DWORDs; pad_interval code 4).
// ---------------------------------------------------------------------------
__device__ __forceinline__ void tdm_load_2d(unsigned ldsOff, const void* gp,
                                            unsigned tensor_d0, unsigned tensor_d1,
                                            unsigned tile_d0, unsigned tile_d1,
                                            unsigned long long stride0) {
  unsigned long long ga = (unsigned long long)gp;   // 57-bit byte address
  v4u g0;
  g0[0] = 1u;                                       // count=1, no gather
  g0[1] = ldsOff;                                   // LDS byte address
  g0[2] = (unsigned)ga;
  g0[3] = (unsigned)(ga >> 32) | (2u << 30);        // type=2 ("image")
  v8i g1;
  // data_size=1 (2B) <<16 | pad_enable<<20 | pad_interval=4 (32 dw) <<22 |
  // pad_amount=3 (4 dw = 16B) <<25 ; workgroup_mask=0
  g1[0] = (int)((1u << 16) | (1u << 20) | (4u << 22) | (3u << 25));
  g1[1] = (int)((tensor_d0 & 0xFFFFu) << 16);            // [47:32]=0, td0 lo
  g1[2] = (int)((tensor_d0 >> 16) | ((tensor_d1 & 0xFFFFu) << 16));
  g1[3] = (int)((tensor_d1 >> 16) | (tile_d0 << 16));
  g1[4] = (int)tile_d1;                                  // tile_dim2 = 0
  g1[5] = (int)(unsigned)stride0;                        // dim0 stride lo
  g1[6] = (int)(unsigned)(stride0 >> 32);                // dim0 stride hi, dim1 stride=0
  g1[7] = 0;
  v4i z4; z4[0] = 0; z4[1] = 0; z4[2] = 0; z4[3] = 0;
  v8i z8;
#pragma unroll
  for (int i = 0; i < 8; ++i) z8[i] = 0;
  __builtin_amdgcn_tensor_load_to_lds(g0, g1, z4, z4, z8, 0);
}

__device__ __forceinline__ unsigned lds_off(const void* p) {
  return (unsigned)(unsigned long long)p;           // LDS offset = low 32 bits
}

// A-fragment (16x32 bf16, MxK): lane = row; low lanes K 0-7 & 16-23,
// high lanes K 8-15 & 24-31 (ISA 7.12.2).
__device__ __forceinline__ v16bf fragA(const unsigned short* rowPtr, int lane) {
  const int c0 = (lane < 16) ? 0 : 8;
  v8bf lo = *(const v8bf*)(rowPtr + c0);
  v8bf hi = *(const v8bf*)(rowPtr + c0 + 16);
  v16bf a;
#pragma unroll
  for (int i = 0; i < 8; ++i) { a[i] = lo[i]; a[i + 8] = hi[i]; }
  return a;
}

// B-fragment (32x16 bf16, KxN) staged as [N][K]: lane = column n; low lanes
// K 0-7 & 16-23, high lanes K 8-15 & 24-31 (same order as fragA).
__device__ __forceinline__ v16bf fragB(const unsigned short* rowPtr, int lane) {
  const int c0 = (lane < 16) ? 0 : 8;
  v8bf lo = *(const v8bf*)(rowPtr + c0);
  v8bf hi = *(const v8bf*)(rowPtr + c0 + 16);
  v16bf b;
#pragma unroll
  for (int i = 0; i < 8; ++i) { b[i] = lo[i]; b[i + 8] = hi[i]; }
  return b;
}

// ---------------------------------------------------------------------------
// GEMM: C[M,N] = A[M,1024]*B[1024,N] (+bias). Block 128x128, 256 thr (8 waves),
// wave tile 32x64 (2x4 WMMA tiles). MODE 0: A=f32 x, scatter q(bf16)/k/v(f32)
// + bf16 k and transposed bf16 v for TDM attention staging. MODE 1: A=bf16 via
// TDM, write f32 y.
// ---------------------------------------------------------------------------
template <int MODE>
__global__ __launch_bounds__(256)
void gemm_kernel(const void* __restrict__ Aptr, const float* __restrict__ Bw,
                 const float* __restrict__ bias,
                 unsigned short* __restrict__ q_ws,
                 float* __restrict__ k_out, float* __restrict__ v_out,
                 unsigned short* __restrict__ kb_ws,
                 unsigned short* __restrict__ vt_ws,
                 float* __restrict__ y_out) {
  __shared__ unsigned short ldsA[128][72];
  __shared__ unsigned short ldsB[128][72];

  const int tid   = threadIdx.x;
  const int lane  = tid & 31;
  const int ln    = lane & 15;
  const int wave  = tid >> 5;
  const int waveM = wave & 3;   // 4 waves along M (32 rows each)
  const int waveN = wave >> 2;  // 2 waves along N (64 cols each)
  const int m0 = blockIdx.y * 128;
  const int n0 = blockIdx.x * 128;
  const int ldb = (MODE == 0) ? 3 * CDIM : CDIM;

  v8f acc[2][4];
#pragma unroll
  for (int mt = 0; mt < 2; ++mt)
#pragma unroll
    for (int nt = 0; nt < 4; ++nt)
#pragma unroll
      for (int i = 0; i < 8; ++i) acc[mt][nt][i] = 0.0f;

  for (int k0 = 0; k0 < 1024; k0 += 64) {
    __syncthreads();
    // ---- stage A tile (128 x 64) ----
    if constexpr (MODE == 0) {
      const float* A = (const float*)Aptr;
#pragma unroll
      for (int it = 0; it < 8; ++it) {
        int idx = it * 256 + tid;            // 0..2047
        int m = idx >> 4, k = (idx & 15) * 4;
        const float4 f = *(const float4*)(A + (size_t)(m0 + m) * 1024 + k0 + k);
        uint2 pk; pk.x = pkbf(f.x, f.y); pk.y = pkbf(f.z, f.w);
        *(uint2*)&ldsA[m][k] = pk;
      }
    } else {
      // bf16 source: DMA the whole tile via the Tensor Data Mover
      if (wave == 0) {
        const unsigned short* A = (const unsigned short*)Aptr;
        tdm_load_2d(lds_off(&ldsA[0][0]), A + (size_t)m0 * 1024 + k0,
                    /*td0=*/1024, /*td1=*/MTOT, /*tile0=*/64, /*tile1=*/128,
                    /*stride0=*/1024ull);
      }
    }
    // ---- stage B tile (64 x 128) transposed -> LDS [N][K] bf16 ----
#pragma unroll
    for (int it = 0; it < 8; ++it) {
      int idx = it * 256 + tid;              // 0..2047
      int kk = idx >> 5, n = (idx & 31) * 4;
      const float4 f = *(const float4*)(Bw + (size_t)(k0 + kk) * ldb + n0 + n);
      unsigned p01 = pkbf(f.x, f.y), p23 = pkbf(f.z, f.w);
      ldsB[n + 0][kk] = (unsigned short)p01;
      ldsB[n + 1][kk] = (unsigned short)(p01 >> 16);
      ldsB[n + 2][kk] = (unsigned short)p23;
      ldsB[n + 3][kk] = (unsigned short)(p23 >> 16);
    }
    if constexpr (MODE == 1) {
      if (wave == 0) __builtin_amdgcn_s_wait_tensorcnt(0);
    }
    __syncthreads();

    // ---- WMMA: 2 K-steps of 32, 2x4 tiles per wave ----
#pragma unroll
    for (int ks = 0; ks < 64; ks += 32) {
      v16bf af[2], bfr[4];
#pragma unroll
      for (int mt = 0; mt < 2; ++mt)
        af[mt] = fragA(&ldsA[waveM * 32 + mt * 16 + ln][ks], lane);
#pragma unroll
      for (int nt = 0; nt < 4; ++nt)
        bfr[nt] = fragB(&ldsB[waveN * 64 + nt * 16 + ln][ks], lane);
#pragma unroll
      for (int mt = 0; mt < 2; ++mt)
#pragma unroll
        for (int nt = 0; nt < 4; ++nt)
          acc[mt][nt] = __builtin_amdgcn_wmma_f32_16x16x32_bf16(
              false, af[mt], false, bfr[nt], (short)0, acc[mt][nt], false, false);
    }
  }

  // ---- epilogue ----
  const int hoff = (lane < 16) ? 0 : 8;
#pragma unroll
  for (int mt = 0; mt < 2; ++mt)
#pragma unroll
    for (int nt = 0; nt < 4; ++nt) {
      int gn = n0 + waveN * 64 + nt * 16 + ln;
      float bv = bias[gn];
      if constexpr (MODE == 0) {
        int which = gn >> 10;              // 0=q 1=k 2=v
        int c = gn & 1023, h = c >> 6, d = c & 63;
#pragma unroll
        for (int i = 0; i < 8; i += 2) {
          int gm = m0 + waveM * 32 + mt * 16 + hoff + i;
          int b = gm >> 11, t = gm & 2047;           // t even
          float v0 = acc[mt][nt][i] + bv;
          float v1 = acc[mt][nt][i + 1] + bv;
          size_t dst = ((size_t)(b * NH + h) * BT + t) * HD + d;
          unsigned p = pkbf(v0, v1);
          if (which == 0) {
            q_ws[dst] = (unsigned short)p;
            q_ws[dst + HD] = (unsigned short)(p >> 16);
          } else if (which == 1) {
            k_out[dst] = v0; k_out[dst + HD] = v1;
            kb_ws[dst] = (unsigned short)p;
            kb_ws[dst + HD] = (unsigned short)(p >> 16);
          } else {
            v_out[dst] = v0; v_out[dst + HD] = v1;
            size_t tdst = ((size_t)(b * NH + h) * HD + d) * BT + t;
            *(unsigned*)&vt_ws[tdst] = p;            // t even -> 4B aligned
          }
        }
      } else {
#pragma unroll
        for (int i = 0; i < 8; ++i) {
          int gm = m0 + waveM * 32 + mt * 16 + hoff + i;
          y_out[(size_t)gm * CDIM + gn] = acc[mt][nt][i] + bv;
        }
      }
    }
}

// ---------------------------------------------------------------------------
// Flash attention: grid (16 q-blocks, 64 b*h). 8 waves x 16 q rows. Key blocks
// of 64; Q/K/Vt staged bf16 via TDM; online softmax; bf16 WMMA.
// ---------------------------------------------------------------------------
__global__ __launch_bounds__(256)
void attn_kernel(const unsigned short* __restrict__ q_ws,
                 const unsigned short* __restrict__ kb_ws,
                 const unsigned short* __restrict__ vt_ws,
                 unsigned short* __restrict__ y_ws) {
  __shared__ unsigned short Qs[128][72];
  __shared__ unsigned short Ks[64][72];
  __shared__ unsigned short Vts[64][72];   // [d][key]
  __shared__ unsigned short Ps[128][72];

  const int tid  = threadIdx.x;
  const int lane = tid & 31;
  const int ln   = lane & 15;
  const int wave = tid >> 5;
  const int hoff = (lane < 16) ? 0 : 8;
  const int qblk = blockIdx.x;
  const int bh   = blockIdx.y;             // b*16 + h
  const size_t baseBH = (size_t)bh * BT * HD;
  const int qBase = qblk * 128;

  // Q tile (128 x 64) via TDM, issued by wave 2
  if (wave == 2) {
    tdm_load_2d(lds_off(&Qs[0][0]), q_ws + baseBH + (size_t)qBase * HD,
                /*td0=*/HD, /*td1=*/BT, /*tile0=*/64, /*tile1=*/128,
                /*stride0=*/(unsigned long long)HD);
    __builtin_amdgcn_s_wait_tensorcnt(0);
  }

  float mrun[8], lrun[8];
  v8f oacc[4];
#pragma unroll
  for (int i = 0; i < 8; ++i) { mrun[i] = -__builtin_huge_valf(); lrun[i] = 0.0f; }
#pragma unroll
  for (int dt = 0; dt < 4; ++dt)
#pragma unroll
    for (int i = 0; i < 8; ++i) oacc[dt][i] = 0.0f;

  const int nKB = 2 * qblk + 2;
  for (int kb = 0; kb < nKB; ++kb) {
    __syncthreads();
    if (wave == 0) {           // K tile [key][d]: 64 x 64
      tdm_load_2d(lds_off(&Ks[0][0]), kb_ws + baseBH + (size_t)(kb * 64) * HD,
                  HD, BT, 64, 64, (unsigned long long)HD);
      __builtin_amdgcn_s_wait_tensorcnt(0);
    } else if (wave == 1) {    // V^T tile [d][key]: 64 x 64 from [B,H,D,T]
      tdm_load_2d(lds_off(&Vts[0][0]), vt_ws + (size_t)bh * HD * BT + kb * 64,
                  BT, HD, 64, 64, (unsigned long long)BT);
      __builtin_amdgcn_s_wait_tensorcnt(0);
    }
    __syncthreads();

    // S = Q K^T : 16 q rows x 64 keys -> 4 tiles
    v8f sacc[4];
#pragma unroll
    for (int nt = 0; nt < 4; ++nt)
#pragma unroll
      for (int i = 0; i < 8; ++i) sacc[nt][i] = 0.0f;
#pragma unroll
    for (int ks = 0; ks < 64; ks += 32) {
      v16bf aq = fragA(&Qs[wave * 16 + ln][ks], lane);
#pragma unroll
      for (int nt = 0; nt < 4; ++nt) {
        v16bf bk = fragB(&Ks[nt * 16 + ln][ks], lane);
        sacc[nt] = __builtin_amdgcn_wmma_f32_16x16x32_bf16(
            false, aq, false, bk, (short)0, sacc[nt], false, false);
      }
    }

    // online softmax
    float sv[4][8];
#pragma unroll
    for (int nt = 0; nt < 4; ++nt) {
      int kcol = kb * 64 + nt * 16 + ln;
#pragma unroll
      for (int i = 0; i < 8; ++i) {
        float s = sacc[nt][i] * 0.125f;
        int qrow = qBase + wave * 16 + hoff + i;
        sv[nt][i] = (kcol > qrow) ? -__builtin_huge_valf() : s;
      }
    }
#pragma unroll
    for (int i = 0; i < 8; ++i) {
      float tm = fmaxf(fmaxf(sv[0][i], sv[1][i]), fmaxf(sv[2][i], sv[3][i]));
#pragma unroll
      for (int off = 8; off >= 1; off >>= 1)
        tm = fmaxf(tm, __shfl_xor(tm, off, 32));
      float mnew = fmaxf(mrun[i], tm);
      float alpha = __expf(mrun[i] - mnew);
      float p0 = __expf(sv[0][i] - mnew);
      float p1 = __expf(sv[1][i] - mnew);
      float p2 = __expf(sv[2][i] - mnew);
      float p3 = __expf(sv[3][i] - mnew);
      float rs = (p0 + p1) + (p2 + p3);
      int prow = wave * 16 + hoff + i;
      unsigned q01 = pkbf(p0, p1), q23 = pkbf(p2, p3);
      Ps[prow][ln]      = (unsigned short)q01;
      Ps[prow][16 + ln] = (unsigned short)(q01 >> 16);
      Ps[prow][32 + ln] = (unsigned short)q23;
      Ps[prow][48 + ln] = (unsigned short)(q23 >> 16);
#pragma unroll
      for (int off = 8; off >= 1; off >>= 1)
        rs += __shfl_xor(rs, off, 32);
      lrun[i] = lrun[i] * alpha + rs;
      mrun[i] = mnew;
#pragma unroll
      for (int dt = 0; dt < 4; ++dt) oacc[dt][i] *= alpha;
    }
    __syncthreads();

    // O += P V
#pragma unroll
    for (int ks = 0; ks < 64; ks += 32) {
      v16bf ap = fragA(&Ps[wave * 16 + ln][ks], lane);
#pragma unroll
      for (int dt = 0; dt < 4; ++dt) {
        v16bf bv = fragB(&Vts[dt * 16 + ln][ks], lane);
        oacc[dt] = __builtin_amdgcn_wmma_f32_16x16x32_bf16(
            false, ap, false, bv, (short)0, oacc[dt], false, false);
      }
    }
  }

  // normalize + store y_att (bf16 [B,T,C])
  const int h = bh & 15, b = bh >> 4;
#pragma unroll
  for (int i = 0; i < 8; i += 2) {
    float inv0 = 1.0f / lrun[i];
    float inv1 = 1.0f / lrun[i + 1];
    int t = qBase + wave * 16 + hoff + i;
#pragma unroll
    for (int dt = 0; dt < 4; ++dt) {
      int c = h * 64 + dt * 16 + ln;
      unsigned p = pkbf(oacc[dt][i] * inv0, oacc[dt][i + 1] * inv1);
      size_t dst = (size_t)(b * BT + t) * CDIM + c;
      y_ws[dst] = (unsigned short)p;
      y_ws[dst + CDIM] = (unsigned short)(p >> 16);
    }
  }
}

extern "C" void kernel_launch(void* const* d_in, const int* in_sizes, int n_in,
                              void* d_out, int out_size, void* d_ws, size_t ws_size,
                              hipStream_t stream) {
  (void)in_sizes; (void)n_in; (void)out_size; (void)ws_size;
  const float* x      = (const float*)d_in[0];
  const float* W_attn = (const float*)d_in[1];
  const float* b_attn = (const float*)d_in[2];
  const float* W_proj = (const float*)d_in[3];
  const float* b_proj = (const float*)d_in[4];

  float* y     = (float*)d_out;                          // [B,T,C]
  float* k_out = y + (size_t)MTOT * CDIM;                // [B,H,T,D]
  float* v_out = k_out + (size_t)MTOT * CDIM;            // [B,H,T,D]

  unsigned short* q_ws    = (unsigned short*)d_ws;            // bf16 [B,H,T,D]
  unsigned short* kb_ws   = q_ws  + (size_t)MTOT * CDIM;      // bf16 [B,H,T,D]
  unsigned short* vt_ws   = kb_ws + (size_t)MTOT * CDIM;      // bf16 [B,H,D,T]
  unsigned short* yatt_ws = vt_ws + (size_t)MTOT * CDIM;      // bf16 [B,T,C]

  // 1) QKV projection (N = 3072)
  gemm_kernel<0><<<dim3(24, 64), 256, 0, stream>>>(
      x, W_attn, b_attn, q_ws, k_out, v_out, kb_ws, vt_ws, nullptr);
  // 2) causal flash attention
  attn_kernel<<<dim3(16, 64), 256, 0, stream>>>(q_ws, kb_ws, vt_ws, yatt_ws);
  // 3) output projection (N = 1024)
  gemm_kernel<1><<<dim3(8, 64), 256, 0, stream>>>(
      yatt_ws, W_proj, b_proj, nullptr, nullptr, nullptr, nullptr, nullptr, y);
}
